// GRNSeq2Seq_24567212933621
// MI455X (gfx1250) — matrix operand order repaired and checked by reading, not compiled, not run on hardware
//
#include <hip/hip_runtime.h>
#include <hip/hip_bf16.h>
#include <stdint.h>

// ---- problem constants (from reference) ----
#define B_    32
#define T_    64
#define N_    500
#define F_    8
#define HG_   16
#define D_    512
#define G3_   1536      // 3*D
#define NHG_  8000      // N*HG (exactly divisible by 32 -> clean WMMA K)
#define BT_   2048      // B*T
#define E_    16000
#define STEPS_ 16
#define NP_   512       // N padded to 512 for 16x16 tiling

typedef __bf16 bf16;
typedef __attribute__((ext_vector_type(16))) __bf16 v16bf;
typedef __attribute__((ext_vector_type(8)))  __bf16 v8bf;
typedef __attribute__((ext_vector_type(8)))  float  v8f;

union AB16 { v16bf v; v8bf h[2]; };

// ---- WMMA operand loaders (layouts per CDNA5 ISA §7.12.2) ----
// A 16x32 bf16: lanes 0-15: row M=lane, K {0..7,16..23}; lanes 16-31: K {8..15,24..31}
__device__ __forceinline__ v16bf load_a_tile(const bf16* M, int ld, int row0, int k0, int lane) {
  const int half = lane >> 4, r = lane & 15;
  const bf16* p = M + (size_t)(row0 + r) * ld + k0 + half * 8;
  AB16 u;
  u.h[0] = *(const v8bf*)(p);
  u.h[1] = *(const v8bf*)(p + 16);
  return u.v;
}

// B 32x16 bf16: lane holds one column (= one row of (O,K)-row-major W), K {0..7,16..23} / {8..15,24..31} like A
__device__ __forceinline__ v16bf load_b_tile(const bf16* Wt, int ld, int o0, int k0, int lane) {
  const int half = lane >> 4, c = lane & 15;
  const bf16* p = Wt + (size_t)(o0 + c) * ld + k0 + half * 8;
  AB16 u;
  u.h[0] = *(const v8bf*)(p);
  u.h[1] = *(const v8bf*)(p + 16);
  return u.v;
}

__device__ __forceinline__ v8f wmma_bf16(v16bf a, v16bf b, v8f c) {
  return __builtin_amdgcn_wmma_f32_16x16x32_bf16(false, a, false, b, (short)0, c, false, false);
}

// 2 row tiles x 4 col tiles per wave, software-pipelined K loop (step 32).
// B tiles shared across the row pair, A tiles shared across the 4 col tiles.
__device__ __forceinline__ void gemm_core24(const bf16* A, int lda, int row0,
                                            const bf16* Bm, int ldb, int o0,
                                            int K, int lane, v8f acc[2][4]) {
  const int c = lane & 15;
  v16bf a0 = load_a_tile(A, lda, row0,      0, lane);
  v16bf a1 = load_a_tile(A, lda, row0 + 16, 0, lane);
  v16bf b0 = load_b_tile(Bm, ldb, o0,      0, lane);
  v16bf b1 = load_b_tile(Bm, ldb, o0 + 16, 0, lane);
  v16bf b2 = load_b_tile(Bm, ldb, o0 + 32, 0, lane);
  v16bf b3 = load_b_tile(Bm, ldb, o0 + 48, 0, lane);
  for (int k0 = 0; k0 < K; k0 += 32) {
    v16bf a0n, a1n, b0n, b1n, b2n, b3n;
    const int kn = k0 + 32;
    if (kn < K) {
      // preload next K-step before consuming current -> loads overlap WMMA
      a0n = load_a_tile(A, lda, row0,      kn, lane);
      a1n = load_a_tile(A, lda, row0 + 16, kn, lane);
      b0n = load_b_tile(Bm, ldb, o0,      kn, lane);
      b1n = load_b_tile(Bm, ldb, o0 + 16, kn, lane);
      b2n = load_b_tile(Bm, ldb, o0 + 32, kn, lane);
      b3n = load_b_tile(Bm, ldb, o0 + 48, kn, lane);
      if (kn + 32 < K)  // stream-ahead hint for the B panel (global_prefetch_b8)
        __builtin_prefetch(Bm + (size_t)(o0 + c) * ldb + kn + 32, 0, 1);
    }
    acc[0][0] = wmma_bf16(a0, b0, acc[0][0]);
    acc[1][0] = wmma_bf16(a1, b0, acc[1][0]);
    acc[0][1] = wmma_bf16(a0, b1, acc[0][1]);
    acc[1][1] = wmma_bf16(a1, b1, acc[1][1]);
    acc[0][2] = wmma_bf16(a0, b2, acc[0][2]);
    acc[1][2] = wmma_bf16(a1, b2, acc[1][2]);
    acc[0][3] = wmma_bf16(a0, b3, acc[0][3]);
    acc[1][3] = wmma_bf16(a1, b3, acc[1][3]);
    a0 = a0n; a1 = a1n;
    b0 = b0n; b1 = b1n; b2 = b2n; b3 = b3n;
  }
}

#define ZERO8 {0.f,0.f,0.f,0.f,0.f,0.f,0.f,0.f}
#define ACCINIT {{ZERO8, ZERO8, ZERO8, ZERO8}, {ZERO8, ZERO8, ZERO8, ZERO8}}

// ---------------- small utility kernels ----------------
__global__ void k_zero(uint32_t* p, int n) {
  int i = blockIdx.x * blockDim.x + threadIdx.x;
  if (i < n) p[i] = 0u;
}
__global__ void k_cvt_bf16(const float* s, bf16* d, int n) {
  int i = blockIdx.x * blockDim.x + threadIdx.x;
  if (i < n) d[i] = (bf16)s[i];
}
__global__ void k_deg(const int* ei, float* deg) {
  int e = blockIdx.x * blockDim.x + threadIdx.x;
  if (e < E_) atomicAdd(&deg[ei[E_ + e]], 1.0f);
}
__global__ void k_dinv(const float* deg, float* dinv) {
  int i = blockIdx.x * blockDim.x + threadIdx.x;
  if (i < N_) dinv[i] = rsqrtf(deg[i] + 1.0f);   // +1 for self loop
}
// dense normalized adjacency: A[dst,src] += dinv[src]*dinv[dst]; self loops on diagonal
__global__ void k_abuild(const int* ei, const float* dinv, float* A) {
  int idx = blockIdx.x * blockDim.x + threadIdx.x;
  if (idx < E_) {
    int s = ei[idx], d = ei[E_ + idx];
    atomicAdd(&A[(size_t)d * NP_ + s], dinv[s] * dinv[d]);
  } else if (idx < E_ + N_) {
    int i = idx - E_;
    atomicAdd(&A[(size_t)i * NP_ + i], dinv[i] * dinv[i]);
  }
}
// encoder GCN feature proj: TmT[(b'*16+c)][j] = bf16(sum_f x[b',j,f]*W[f,c])
__global__ void k_tm1(const float* x, const float* gW, bf16* TmT) {
  int id = blockIdx.x * blockDim.x + threadIdx.x;
  if (id >= BT_ * N_ * HG_) return;
  int c  = id & 15;
  int j  = (id >> 4) % N_;
  int bp = id / (N_ * HG_);
  const float* xr = x + ((size_t)bp * N_ + j) * F_;
  float v = 0.f;
#pragma unroll
  for (int f = 0; f < F_; ++f) v += xr[f] * gW[f * HG_ + c];
  TmT[(size_t)(bp * 16 + c) * NP_ + j] = (bf16)v;
}
// decoder GCN proj: TmT[(b*16+c)][j] = bf16(inp[b,j] * Wdec[c])
__global__ void k_tmdec(const float* inp, const float* gW, bf16* TmT) {
  int id = blockIdx.x * blockDim.x + threadIdx.x;
  if (id >= B_ * N_ * HG_) return;
  int c = id & 15;
  int j = (id >> 4) % N_;
  int b = id / (N_ * HG_);
  float v = inp[b * N_ + j] * gW[c];
  TmT[(size_t)(b * 16 + c) * NP_ + j] = (bf16)v;
}

// ---------------- GEMM kernels (WMMA bf16 -> f32) ----------------

// encoder GCN aggregate: C(512x32768) = A(512x512) @ TmT^T ; epilogue relu+bias,
// scatter into Xenc[(t*B+b)][n*16+c] (bf16, A-operand layout for the GI GEMM)
__global__ void __launch_bounds__(256) k_gemm_gcn_enc(const bf16* Abf, const bf16* TmT,
                                                      const float* gcnb, bf16* Xenc) {
  int wid = blockIdx.x * 8 + (threadIdx.x >> 5);
  int lane = threadIdx.x & 31;
  int rp = wid >> 9;          // 0..15  node row-pair tile (32 rows)
  int cg = wid & 511;         // 0..511 col group of 4 tiles
  v8f acc[2][4] = ACCINIT;
  gemm_core24(Abf, NP_, rp * 32, TmT, NP_, cg * 64, NP_, lane, acc);
  int half = lane >> 4, col = lane & 15;
  float bb = gcnb[col];
#pragma unroll
  for (int t = 0; t < 4; ++t) {
    int bp = cg * 4 + t;                 // b' = b*T + tt
    int b = bp >> 6, tt = bp & 63;
    bf16* xr = Xenc + (size_t)(tt * B_ + b) * NHG_;
#pragma unroll
    for (int i = 0; i < 2; ++i) {
#pragma unroll
      for (int r = 0; r < 8; ++r) {
        int n = rp * 32 + i * 16 + r + half * 8;
        if (n < N_) {
          float v = acc[i][t][r] + bb;
          v = v > 0.f ? v : 0.f;
          xr[n * 16 + col] = (bf16)v;
        }
      }
    }
  }
}

// GI = X(rows x 8000) @ Wih^T + bih  -> f32 (rows x 1536). Used by encoder (rows=2048,
// 64 row pairs) and decoder (rows=32, 1 row pair); grid size selects rp range.
__global__ void __launch_bounds__(256) k_gemm_gi(const bf16* X, const bf16* Wih,
                                                 const float* bih, float* GI) {
  int wid = blockIdx.x * 8 + (threadIdx.x >> 5);
  int lane = threadIdx.x & 31;
  int rp = wid / 24;
  int cg = wid % 24;
  int o0 = cg * 64;
  v8f acc[2][4] = ACCINIT;
  gemm_core24(X, NHG_, rp * 32, Wih, NHG_, o0, NHG_, lane, acc);
  int half = lane >> 4, col = lane & 15;
#pragma unroll
  for (int t = 0; t < 4; ++t) {
    int o = o0 + t * 16 + col;
    float bb = bih[o];
#pragma unroll
    for (int i = 0; i < 2; ++i) {
#pragma unroll
      for (int r = 0; r < 8; ++r) {
        int row = rp * 32 + i * 16 + r + half * 8;
        GI[(size_t)row * G3_ + o] = acc[i][t][r] + bb;
      }
    }
  }
}

// decoder GCN aggregate: C(512x512) = A @ TmdecT^T; relu+bias -> Xdec[b][n*16+c]
__global__ void __launch_bounds__(256) k_gemm_gcn_dec(const bf16* Abf, const bf16* TmT,
                                                      const float* gcnb, bf16* Xdec) {
  int wid = blockIdx.x * 8 + (threadIdx.x >> 5);
  int lane = threadIdx.x & 31;
  int rp = wid >> 3;          // 0..15
  int cg = wid & 7;           // 0..7
  v8f acc[2][4] = ACCINIT;
  gemm_core24(Abf, NP_, rp * 32, TmT, NP_, cg * 64, NP_, lane, acc);
  int half = lane >> 4, col = lane & 15;
  float bb = gcnb[col];
#pragma unroll
  for (int t = 0; t < 4; ++t) {
    int b = cg * 4 + t;       // 0..31
#pragma unroll
    for (int i = 0; i < 2; ++i) {
#pragma unroll
      for (int r = 0; r < 8; ++r) {
        int n = rp * 32 + i * 16 + r + half * 8;
        if (n < N_) {
          float v = acc[i][t][r] + bb;
          v = v > 0.f ? v : 0.f;
          Xdec[(size_t)b * NHG_ + n * 16 + col] = (bf16)v;
        }
      }
    }
  }
}

// FC: out(32x500) = h(32x512) @ fcW^T + fcb ; writes d_out[:,s,:] and next decoder input
__global__ void __launch_bounds__(256) k_fc(const bf16* hgb, const bf16* fcw, const float* fcb,
                                            float* out, float* ginp, int s) {
  int wid = blockIdx.x * 8 + (threadIdx.x >> 5);
  int lane = threadIdx.x & 31;
  int cg = wid & 7;           // 0..7 (single row pair: all 32 batch rows)
  int o0 = cg * 64;
  v8f acc[2][4] = ACCINIT;
  gemm_core24(hgb, D_, 0, fcw, D_, o0, D_, lane, acc);
  int half = lane >> 4, col = lane & 15;
#pragma unroll
  for (int t = 0; t < 4; ++t) {
    int n = o0 + t * 16 + col;
    if (n < N_) {
      float bb = fcb[n];
#pragma unroll
      for (int i = 0; i < 2; ++i) {
#pragma unroll
        for (int r = 0; r < 8; ++r) {
          int b = i * 16 + r + half * 8;   // batch row
          float v = acc[i][t][r] + bb;
          out[(size_t)b * (STEPS_ * N_) + s * N_ + n] = v;
          ginp[b * N_ + n] = v;
        }
      }
    }
  }
}

// ---------------- fused GRU recurrence: one workgroup, h lives in LDS ----------------
// 32 waves; each wave owns 2 (row-tile, d-tile) pairs => computes the r/z/n gh tiles
// for its 16x16 (batch, d) patch and applies the GRU update in-register.
__global__ void __launch_bounds__(1024) k_gru(const float* gi, const bf16* whh, const float* bhh,
                                              float* hg, bf16* hgb, int nsteps, int init_from_global) {
  extern __shared__ char smem[];
  float* hs = (float*)smem;                       // 32*512 f32  (64 KB)
  bf16*  hb = (bf16*)(smem + B_ * D_ * 4);        // 32*512 bf16 (32 KB)
  const int tid = threadIdx.x;
  const int wave = tid >> 5, lane = tid & 31;
  const int half = lane >> 4, col = lane & 15;

  for (int i = tid; i < B_ * D_; i += 1024) {
    float v = init_from_global ? hg[i] : 0.f;
    hs[i] = v;
    hb[i] = (bf16)v;
  }
  __syncthreads();

  for (int t = 0; t < nsteps; ++t) {
    v8f cr[2] = {ZERO8, ZERO8}, cz[2] = {ZERO8, ZERO8}, cn[2] = {ZERO8, ZERO8};
#pragma unroll
    for (int pp = 0; pp < 2; ++pp) {
      int pid = wave * 2 + pp;
      int rt = pid & 1;          // batch row tile (0..1)
      int j  = pid >> 1;         // d tile (0..31)
      for (int k0 = 0; k0 < D_; k0 += 32) {
        v16bf a = load_a_tile(hb, D_, rt * 16, k0, lane);     // LDS -> ds_load
        cr[pp] = wmma_bf16(a, load_b_tile(whh, D_,            j * 16, k0, lane), cr[pp]);
        cz[pp] = wmma_bf16(a, load_b_tile(whh, D_, D_       + j * 16, k0, lane), cz[pp]);
        cn[pp] = wmma_bf16(a, load_b_tile(whh, D_, 2 * D_   + j * 16, k0, lane), cn[pp]);
      }
    }
    __syncthreads();   // all waves done reading h (bf16) for this step
#pragma unroll
    for (int pp = 0; pp < 2; ++pp) {
      int pid = wave * 2 + pp;
      int rt = pid & 1;
      int j  = pid >> 1;
      int o  = j * 16 + col;
      float br = bhh[o], bz = bhh[D_ + o], bn = bhh[2 * D_ + o];
#pragma unroll
      for (int r = 0; r < 8; ++r) {
        int row = rt * 16 + r + half * 8;
        const float* gir = gi + (size_t)(t * B_ + row) * G3_;
        float rr = 1.f / (1.f + __expf(-(gir[o]          + cr[pp][r] + br)));
        float zz = 1.f / (1.f + __expf(-(gir[D_ + o]     + cz[pp][r] + bz)));
        float nn = tanhf(        gir[2 * D_ + o] + rr * (cn[pp][r] + bn));
        int idx = row * D_ + o;
        float ho = hs[idx];
        float hn = (1.f - zz) * nn + zz * ho;
        hs[idx] = hn;
        hb[idx] = (bf16)hn;
      }
    }
    __syncthreads();   // h updated before next step's GEMM reads
  }

  for (int i = tid; i < B_ * D_; i += 1024) {
    hg[i]  = hs[i];
    hgb[i] = hb[i];
  }
}

// ---------------- host orchestration ----------------
extern "C" void kernel_launch(void* const* d_in, const int* in_sizes, int n_in,
                              void* d_out, int out_size, void* d_ws, size_t ws_size,
                              hipStream_t stream) {
  (void)in_sizes; (void)n_in; (void)out_size; (void)ws_size;
  const float* x        = (const float*)d_in[0];
  const float* dec_init = (const float*)d_in[1];
  const int*   ei       = (const int*)  d_in[2];
  const float* gcn_eW   = (const float*)d_in[3];
  const float* gcn_eb   = (const float*)d_in[4];
  const float* gcn_dW   = (const float*)d_in[5];
  const float* gcn_db   = (const float*)d_in[6];
  const float* enc_Wih  = (const float*)d_in[7];
  const float* enc_Whh  = (const float*)d_in[8];
  const float* enc_bih  = (const float*)d_in[9];
  const float* enc_bhh  = (const float*)d_in[10];
  const float* dec_Wih  = (const float*)d_in[11];
  const float* dec_Whh  = (const float*)d_in[12];
  const float* dec_bih  = (const float*)d_in[13];
  const float* dec_bhh  = (const float*)d_in[14];
  const float* fc_W     = (const float*)d_in[15];
  const float* fc_b     = (const float*)d_in[16];
  float* out = (float*)d_out;

  // workspace carve-up (256B aligned)
  char* w = (char*)d_ws;
  auto alloc = [&](size_t bytes) -> char* {
    char* p = w;
    w += (bytes + 255) & ~(size_t)255;
    return p;
  };
  float* deg   = (float*)alloc(NP_ * 4);
  float* dinv  = (float*)alloc(NP_ * 4);
  float* Af    = (float*)alloc((size_t)NP_ * NP_ * 4);
  bf16*  Abf   = (bf16*) alloc((size_t)NP_ * NP_ * 2);
  bf16*  WihE  = (bf16*) alloc((size_t)G3_ * NHG_ * 2);
  bf16*  WihD  = (bf16*) alloc((size_t)G3_ * NHG_ * 2);
  bf16*  WhhE  = (bf16*) alloc((size_t)G3_ * D_ * 2);
  bf16*  WhhD  = (bf16*) alloc((size_t)G3_ * D_ * 2);
  bf16*  fcWb  = (bf16*) alloc((size_t)NP_ * D_ * 2);
  bf16*  Tm1T  = (bf16*) alloc((size_t)BT_ * HG_ * NP_ * 2);   // (32768 x 512)
  bf16*  Xenc  = (bf16*) alloc((size_t)BT_ * NHG_ * 2);        // (2048 x 8000)
  float* GIenc = (float*)alloc((size_t)BT_ * G3_ * 4);         // (2048 x 1536)
  bf16*  TmdT  = (bf16*) alloc((size_t)NP_ * NP_ * 2);         // (512 x 512)
  bf16*  Xdec  = (bf16*) alloc((size_t)B_ * NHG_ * 2);         // (32 x 8000)
  float* GIdec = (float*)alloc((size_t)B_ * G3_ * 4);          // (32 x 1536)
  float* hg    = (float*)alloc((size_t)B_ * D_ * 4);
  bf16*  hgb   = (bf16*) alloc((size_t)B_ * D_ * 2);
  float* ginp  = (float*)alloc((size_t)B_ * N_ * 4);

  auto grid1 = [](int n) { return dim3((n + 255) / 256); };

  // 0) zero buffers whose padding must be zero
  k_zero<<<grid1(NP_),                 256, 0, stream>>>((uint32_t*)deg, NP_);
  k_zero<<<grid1(NP_ * NP_),           256, 0, stream>>>((uint32_t*)Af, NP_ * NP_);
  k_zero<<<grid1(BT_ * HG_ * NP_ / 2), 256, 0, stream>>>((uint32_t*)Tm1T, BT_ * HG_ * NP_ / 2);
  k_zero<<<grid1(NP_ * NP_ / 2),       256, 0, stream>>>((uint32_t*)TmdT, NP_ * NP_ / 2);
  k_zero<<<grid1(NP_ * D_ / 2),        256, 0, stream>>>((uint32_t*)fcWb, NP_ * D_ / 2);

  // 1) graph normalization + dense adjacency
  k_deg   <<<grid1(E_),        256, 0, stream>>>(ei, deg);
  k_dinv  <<<grid1(N_),        256, 0, stream>>>(deg, dinv);
  k_abuild<<<grid1(E_ + N_),   256, 0, stream>>>(ei, dinv, Af);
  k_cvt_bf16<<<grid1(NP_ * NP_), 256, 0, stream>>>(Af, Abf, NP_ * NP_);

  // 2) weight conversions (f32 -> bf16, layout preserved: rows = O, K contiguous)
  k_cvt_bf16<<<grid1(G3_ * NHG_), 256, 0, stream>>>(enc_Wih, WihE, G3_ * NHG_);
  k_cvt_bf16<<<grid1(G3_ * NHG_), 256, 0, stream>>>(dec_Wih, WihD, G3_ * NHG_);
  k_cvt_bf16<<<grid1(G3_ * D_),   256, 0, stream>>>(enc_Whh, WhhE, G3_ * D_);
  k_cvt_bf16<<<grid1(G3_ * D_),   256, 0, stream>>>(dec_Whh, WhhD, G3_ * D_);
  k_cvt_bf16<<<grid1(N_ * D_),    256, 0, stream>>>(fc_W,    fcWb, N_ * D_);

  // 3) encoder: GCN proj -> GCN aggregate (GEMM) -> batched GI GEMM -> fused GRU scan
  k_tm1<<<grid1(BT_ * N_ * HG_), 256, 0, stream>>>(x, gcn_eW, Tm1T);
  k_gemm_gcn_enc<<<dim3(1024), 256, 0, stream>>>(Abf, Tm1T, gcn_eb, Xenc);    // 8192 waves
  k_gemm_gi     <<<dim3(192),  256, 0, stream>>>(Xenc, WihE, enc_bih, GIenc); // 1536 waves
  k_gru<<<dim3(1), dim3(1024), 96 * 1024, stream>>>(GIenc, WhhE, enc_bhh, hg, hgb, T_, 0);

  // 4) autoregressive decoder
  for (int s = 0; s < STEPS_; ++s) {
    const float* inp = (s == 0) ? dec_init : ginp;
    k_tmdec<<<grid1(B_ * N_ * HG_), 256, 0, stream>>>(inp, gcn_dW, TmdT);
    k_gemm_gcn_dec<<<dim3(16), 256, 0, stream>>>(Abf, TmdT, gcn_db, Xdec);    // 128 waves
    k_gemm_gi     <<<dim3(3),  256, 0, stream>>>(Xdec, WihD, dec_bih, GIdec); // 24 waves
    k_gru<<<dim3(1), dim3(1024), 96 * 1024, stream>>>(GIdec, WhhD, dec_bhh, hg, hgb, 1, 1);
    k_fc<<<dim3(1), 256, 0, stream>>>(hgb, fcWb, fc_b, out, ginp, s);         // 8 waves
  }
}
